// EdgeConvLayer_26963804684930
// MI455X (gfx1250) — matrix operand order repaired and checked by reading, not compiled, not run on hardware
//
#include <hip/hip_runtime.h>
#include <hip/hip_bf16.h>

// EdgeConv layer for MI455X (gfx1250, wave32, WMMA).
// gather -> GEMM1(f16 WMMA) -> BN1 stats -> BN1+ReLU fused into GEMM2(f16
// WMMA) -> BN2 stats -> BN2+ReLU+atomic scatter-add.
// bias before training-mode BatchNorm cancels exactly, so b1/b2 unused.
// Per-wave WMMA strip: 32x64 (2 M-tiles x 4 N-tiles). Fragment loads are
// pipelined one K-step ahead of the WMMAs; sched_group_barrier directives
// (DS-read x12, WMMA x8 per K-step) pin the schedule so the backend cannot
// sink the loads back to just-in-time.

typedef __attribute__((ext_vector_type(16))) _Float16 v16h;
typedef __attribute__((ext_vector_type(8)))  _Float16 h8;
typedef __attribute__((ext_vector_type(4)))  _Float16 h4;
typedef __attribute__((ext_vector_type(8)))  float    v8f;

#define D 128
#define BN_EPS 1e-5f

#define SG_DSREAD 0x100
#define SG_WMMA   0x008

__device__ __forceinline__ v8f wmma16(v16h a, v16h b, v8f c) {
  // 8 args: (neg_a, A, neg_b, B, c_mod, C, reuse_a, reuse_b)
  return __builtin_amdgcn_wmma_f32_16x16x32_f16(false, a, false, b, (short)0, c,
                                                false, false);
}

__device__ __forceinline__ void atomAddF(float* p, float v) {
  __hip_atomic_fetch_add(p, v, __ATOMIC_RELAXED, __HIP_MEMORY_SCOPE_AGENT);
}

__device__ __forceinline__ v16h frag_from(const _Float16* p, int off0,
                                          int off1) {
  h8 lo = *(const h8*)(p + off0);
  h8 hi = *(const h8*)(p + off1);
  v16h r;
#pragma unroll
  for (int i = 0; i < 8; i++) { r[i] = lo[i]; r[i + 8] = hi[i]; }
  return r;
}

// Shared inner loop: 2 M-tiles x 4 N-tiles, K-step 32, software-pipelined
// one K-step ahead; schedule pinned via sched_group_barrier.
template <int K, int WSTRIDE>
__device__ __forceinline__ void mma_strip(const _Float16* ar0,
                                          const _Float16* ar1,
                                          const _Float16* br0, int lh,
                                          v8f acc[2][4]) {
  v16h a0c, a1c, bc[4];
  {
    int ka = lh * 8, kb = lh * 8;
    a0c = frag_from(ar0, ka, ka + 16);
    a1c = frag_from(ar1, ka, ka + 16);
#pragma unroll
    for (int t = 0; t < 4; t++)
      bc[t] = frag_from(br0 + t * 16 * WSTRIDE, kb, kb + 16);
  }
  // group 0: the 12 prologue DS reads (L0)
  __builtin_amdgcn_sched_group_barrier(SG_DSREAD, 12, 0);
#pragma unroll
  for (int k0 = 0; k0 < K; k0 += 32) {
    v16h a0n, a1n, bn[4];
    if (k0 + 32 < K) {  // prefetch next K-step's fragments
      int ka = k0 + 32 + lh * 8, kb = k0 + 32 + lh * 8;
      a0n = frag_from(ar0, ka, ka + 16);
      a1n = frag_from(ar1, ka, ka + 16);
#pragma unroll
      for (int t = 0; t < 4; t++)
        bn[t] = frag_from(br0 + t * 16 * WSTRIDE, kb, kb + 16);
    }
#pragma unroll
    for (int t = 0; t < 4; t++) {
      acc[0][t] = wmma16(a0c, bc[t], acc[0][t]);
      acc[1][t] = wmma16(a1c, bc[t], acc[1][t]);
    }
    if (k0 + 32 < K) {
      a0c = a0n;
      a1c = a1n;
#pragma unroll
      for (int t = 0; t < 4; t++) bc[t] = bn[t];
    }
    // per K-step groups: L(i+1) loads first, then the 8 WMMAs of step i
    __builtin_amdgcn_sched_group_barrier(SG_DSREAD, 12, 0);
    __builtin_amdgcn_sched_group_barrier(SG_WMMA, 8, 0);
  }
}

// --------------------------------------------------------------- f32 -> f16
__global__ __launch_bounds__(256) void k_f32tof16(const float* __restrict__ s,
                                                  _Float16* __restrict__ d,
                                                  int n) {
  int i = blockIdx.x * 256 + threadIdx.x;
  if (i < n) d[i] = (_Float16)s[i];
}

// ------------------------------------------------- GEMM1: (E,256)x(256,128)
// 128 threads = 4 waves; 64 edges per block.
// wave w: edges [(w>>1)*32, +32) x features [(w&1)*64, +64).
__global__ __launch_bounds__(128) void k_gemm1(
    const _Float16* __restrict__ xh, const int* __restrict__ ei,
    const _Float16* __restrict__ W1h, _Float16* __restrict__ h1,
    float* __restrict__ psum, float* __restrict__ pss, int E) {
  __shared__ _Float16 As[64][264];   // edge features (f16), padded
  __shared__ _Float16 Ws[128][264];  // W1 row-major (out,in)
  __shared__ _Float16 Hs[64][136];   // raw GEMM output (pre-BN)

  const int tid = threadIdx.x;
  const int e0  = blockIdx.x * 64;

  // W1 -> LDS: thread t owns output-feature row t (256 halves)
  {
    const h8* src = (const h8*)(W1h + tid * 256);
#pragma unroll
    for (int c = 0; c < 32; c++) *(h8*)&Ws[tid][c * 8] = src[c];
  }
  // Gather (f16 source): thread t -> edge r=t/2, side s=t&1, 128 halves
  {
    int r = tid >> 1, s = tid & 1;
    int e = e0 + r; if (e >= E) e = E - 1;
    int idx = ei[s * E + e];
    const h8* src = (const h8*)(xh + (size_t)idx * D);
#pragma unroll
    for (int c = 0; c < 16; c++) *(h8*)&As[r][s * 128 + c * 8] = src[c];
  }
  __syncthreads();

  const int wv = tid >> 5, lane = tid & 31;
  const int l16 = lane & 15, lh = lane >> 4;
  const int m_base = (wv >> 1) * 32;
  const int n_base = (wv & 1) * 64;

  v8f acc[2][4];
#pragma unroll
  for (int mi = 0; mi < 2; mi++)
#pragma unroll
    for (int t = 0; t < 4; t++)
#pragma unroll
      for (int i = 0; i < 8; i++) acc[mi][t][i] = 0.f;

  mma_strip<256, 264>(&As[m_base + l16][0], &As[m_base + 16 + l16][0],
                      &Ws[n_base + l16][0], lh, acc);

  // C/D layout: VGPR v -> M = v + 8*lh, N = l16
#pragma unroll
  for (int mi = 0; mi < 2; mi++)
#pragma unroll
    for (int t = 0; t < 4; t++) {
      int col = n_base + t * 16 + l16;
#pragma unroll
      for (int v = 0; v < 8; v++) {
        int m = m_base + mi * 16 + lh * 8 + v;
        Hs[m][col] = (e0 + m < E) ? (_Float16)acc[mi][t][v] : (_Float16)0.f;
      }
    }
  __syncthreads();

  // store h1 (raw, pre-BN), coalesced 16B chunks
#pragma unroll
  for (int i = 0; i < 8; i++) {
    int lin = i * 1024 + tid * 8;
    int r = lin >> 7, c = lin & 127;
    if (e0 + r < E)
      *(h8*)(h1 + (size_t)(e0 + r) * D + c) = *(const h8*)&Hs[r][c];
  }
  // per-feature partial sums for BN1 batch stats
  {
    float s = 0.f, ss = 0.f;
#pragma unroll 8
    for (int r = 0; r < 64; r++) {
      float v = (float)Hs[r][tid];
      s += v; ss += v * v;
    }
    int row = blockIdx.x & 255;
    atomAddF(&psum[row * D + tid], s);
    atomAddF(&pss[row * D + tid], ss);
  }
}

// ------------------------------------------- finalize BN: a=g*istd, c=be-m*a
__global__ __launch_bounds__(128) void k_finalize(
    const float* __restrict__ psum, const float* __restrict__ pss,
    const float* __restrict__ g, const float* __restrict__ be,
    float* __restrict__ a, float* __restrict__ c, float invE) {
  int f = threadIdx.x;
  float s = 0.f, ss = 0.f;
  for (int r = 0; r < 256; r++) { s += psum[r * D + f]; ss += pss[r * D + f]; }
  float mean = s * invE;
  float var  = ss * invE - mean * mean;
  float istd = rsqrtf(var + BN_EPS);
  float av   = istd * g[f];
  a[f] = av;
  c[f] = be[f] - mean * av;
}

// ----------------------------- GEMM2: relu(bn1(h1)) (E,128) x W2^T (128,128)
__global__ __launch_bounds__(128) void k_gemm2(
    const _Float16* __restrict__ h1, const _Float16* __restrict__ W2h,
    const float* __restrict__ a1, const float* __restrict__ c1,
    _Float16* __restrict__ h2, float* __restrict__ psum,
    float* __restrict__ pss, int E) {
  __shared__ _Float16 As[64][136];
  __shared__ _Float16 Ws[128][136];
  __shared__ _Float16 Hs[64][136];
  __shared__ float a1s[128], c1s[128];

  const int tid = threadIdx.x;
  const int e0  = blockIdx.x * 64;

  a1s[tid] = a1[tid];
  c1s[tid] = c1[tid];
  {
    const h8* src = (const h8*)(W2h + tid * 128);
#pragma unroll
    for (int c = 0; c < 16; c++) *(h8*)&Ws[tid][c * 8] = src[c];
  }
  __syncthreads();
  // load h1 tile, apply BN1 + ReLU on the fly, store f16 A-tile
  {
    int r = tid >> 1, s = tid & 1;
    int e = e0 + r; if (e >= E) e = E - 1;
    const h8* src = (const h8*)(h1 + (size_t)e * D + s * 64);
#pragma unroll
    for (int c = 0; c < 8; c++) {
      h8 hv = src[c];
      h8 ov;
#pragma unroll
      for (int j = 0; j < 8; j++) {
        int f = s * 64 + c * 8 + j;
        float v = fmaf((float)hv[j], a1s[f], c1s[f]);
        ov[j] = (_Float16)fmaxf(v, 0.f);
      }
      *(h8*)&As[r][s * 64 + c * 8] = ov;
    }
  }
  __syncthreads();

  const int wv = tid >> 5, lane = tid & 31;
  const int l16 = lane & 15, lh = lane >> 4;
  const int m_base = (wv >> 1) * 32;
  const int n_base = (wv & 1) * 64;

  v8f acc[2][4];
#pragma unroll
  for (int mi = 0; mi < 2; mi++)
#pragma unroll
    for (int t = 0; t < 4; t++)
#pragma unroll
      for (int i = 0; i < 8; i++) acc[mi][t][i] = 0.f;

  mma_strip<128, 136>(&As[m_base + l16][0], &As[m_base + 16 + l16][0],
                      &Ws[n_base + l16][0], lh, acc);

#pragma unroll
  for (int mi = 0; mi < 2; mi++)
#pragma unroll
    for (int t = 0; t < 4; t++) {
      int col = n_base + t * 16 + l16;
#pragma unroll
      for (int v = 0; v < 8; v++) {
        int m = m_base + mi * 16 + lh * 8 + v;
        Hs[m][col] = (e0 + m < E) ? (_Float16)acc[mi][t][v] : (_Float16)0.f;
      }
    }
  __syncthreads();

#pragma unroll
  for (int i = 0; i < 8; i++) {
    int lin = i * 1024 + tid * 8;
    int r = lin >> 7, c = lin & 127;
    if (e0 + r < E)
      *(h8*)(h2 + (size_t)(e0 + r) * D + c) = *(const h8*)&Hs[r][c];
  }
  {
    float s = 0.f, ss = 0.f;
#pragma unroll 8
    for (int r = 0; r < 64; r++) {
      float v = (float)Hs[r][tid];
      s += v; ss += v * v;
    }
    int row = blockIdx.x & 255;
    atomAddF(&psum[row * D + tid], s);
    atomAddF(&pss[row * D + tid], ss);
  }
}

// ---------------------------------------- BN2 + ReLU + scatter-add by dst
__global__ __launch_bounds__(256) void k_scatter(
    const _Float16* __restrict__ h2, const int* __restrict__ ei,
    const float* __restrict__ a2, const float* __restrict__ c2,
    float* __restrict__ out, int E) {
  long gid = (long)blockIdx.x * 256 + threadIdx.x;
  int e = (int)(gid >> 5);
  if (e >= E) return;
  int q  = (int)(gid & 31);
  int f0 = q * 4;
  int d  = ei[E + e];  // dst row of edge_index
  h4 hv  = *(const h4*)(h2 + (size_t)e * D + f0);
  float* o = out + (size_t)d * D + f0;
#pragma unroll
  for (int j = 0; j < 4; j++) {
    float v = fmaf((float)hv[j], a2[f0 + j], c2[f0 + j]);
    atomAddF(o + j, fmaxf(v, 0.f));
  }
}

// -----------------------------------------------------------------------
extern "C" void kernel_launch(void* const* d_in, const int* in_sizes, int n_in,
                              void* d_out, int out_size, void* d_ws,
                              size_t ws_size, hipStream_t stream) {
  const float* x   = (const float*)d_in[0];
  const int*   ei  = (const int*)d_in[1];
  const float* W1  = (const float*)d_in[2];
  // d_in[3] = b1 : unused (bias cancels under training-mode BN)
  const float* g1  = (const float*)d_in[4];
  const float* be1 = (const float*)d_in[5];
  const float* W2  = (const float*)d_in[6];
  // d_in[7] = b2 : unused
  const float* g2  = (const float*)d_in[8];
  const float* be2 = (const float*)d_in[9];
  float* out = (float*)d_out;

  const int NX = in_sizes[0];      // N * 128
  const int E  = in_sizes[1] / 2;

  char* ws = (char*)d_ws;
  size_t off = 0;
  auto alloc = [&](size_t bytes) -> void* {
    void* p = ws + off;
    off += (bytes + 255) & ~(size_t)255;
    return p;
  };
  _Float16* h1  = (_Float16*)alloc((size_t)E * D * sizeof(_Float16));
  _Float16* h2  = (_Float16*)alloc((size_t)E * D * sizeof(_Float16));
  _Float16* xh  = (_Float16*)alloc((size_t)NX * sizeof(_Float16));
  _Float16* W1h = (_Float16*)alloc(128 * 256 * sizeof(_Float16));
  _Float16* W2h = (_Float16*)alloc(128 * 128 * sizeof(_Float16));
  float* psum1  = (float*)alloc(256 * D * sizeof(float));
  float* pss1   = (float*)alloc(256 * D * sizeof(float));
  float* psum2  = (float*)alloc(256 * D * sizeof(float));
  float* pss2   = (float*)alloc(256 * D * sizeof(float));
  float* a1     = (float*)alloc(D * sizeof(float));
  float* c1     = (float*)alloc(D * sizeof(float));
  float* a2     = (float*)alloc(D * sizeof(float));
  float* c2     = (float*)alloc(D * sizeof(float));

  hipMemsetAsync(d_out, 0, (size_t)out_size * sizeof(float), stream);
  hipMemsetAsync(psum1, 0, 2 * 256 * D * sizeof(float), stream);  // psum1+pss1
  hipMemsetAsync(psum2, 0, 2 * 256 * D * sizeof(float), stream);  // psum2+pss2

  const int nb = (E + 63) / 64;
  const float invE = 1.0f / (float)E;

  k_f32tof16<<<(NX + 255) / 256, 256, 0, stream>>>(x, xh, NX);
  k_f32tof16<<<(128 * 256 + 255) / 256, 256, 0, stream>>>(W1, W1h, 128 * 256);
  k_f32tof16<<<(128 * 128 + 255) / 256, 256, 0, stream>>>(W2, W2h, 128 * 128);

  k_gemm1<<<nb, 128, 0, stream>>>(xh, ei, W1h, h1, psum1, pss1, E);
  k_finalize<<<1, 128, 0, stream>>>(psum1, pss1, g1, be1, a1, c1, invE);
  k_gemm2<<<nb, 128, 0, stream>>>(h1, W2h, a1, c1, h2, psum2, pss2, E);
  k_finalize<<<1, 128, 0, stream>>>(psum2, pss2, g2, be2, a2, c2, invE);

  long sthreads = (long)E * 32;
  k_scatter<<<(int)((sthreads + 255) / 256), 256, 0, stream>>>(h2, ei, a2, c2,
                                                               out, E);
}
